// ComplexDotProductAttention_27736898798432
// MI455X (gfx1250) — compile-verified, matrix operand order checked by reading, not run on hardware
//
#include <hip/hip_runtime.h>
#include <math.h>

typedef __attribute__((ext_vector_type(16))) _Float16 v16h;
typedef __attribute__((ext_vector_type(8)))  _Float16 v8h;
typedef __attribute__((ext_vector_type(4)))  _Float16 v4h;
typedef __attribute__((ext_vector_type(8)))  float    v8f;

#define NB    4      // batches
#define SS    4096   // sequence length
#define ROW   128    // D*2 elements per token row (interleaved r,i)
#define KT    128    // key tile (4 x K=32 WMMA contraction chunks)
#define WAVES 8
#define QW    16     // query rows per wave
#define NTILE (SS / KT)

__device__ __forceinline__ float fast_exp2(float x) {
#if __has_builtin(__builtin_amdgcn_exp2f)
    return __builtin_amdgcn_exp2f(x);   // raw v_exp_f32
#else
    return exp2f(x);
#endif
}

__device__ __forceinline__ v8f wmma_f16(const v16h& a, const v16h& b, const v8f& c) {
    return __builtin_amdgcn_wmma_f32_16x16x32_f16(false, a, false, b, (short)0, c, false, false);
}

// Load a v16h operand: two contiguous 8-half runs (base, base+16 halves),
// each one ds_load_b128.
__device__ __forceinline__ v16h ld_op(const _Float16* base) {
    v8h lo = *(const v8h*)(base);
    v8h hi = *(const v8h*)(base + 16);
    return __builtin_shufflevector(lo, hi, 0,1,2,3,4,5,6,7,8,9,10,11,12,13,14,15);
}

// ---- pre-pass 1: K (f32, interleaved complex) -> two f16 variants in ws
__global__ __launch_bounds__(256)
void prep_k_kernel(const float* __restrict__ K,
                   _Float16* __restrict__ Ksr, _Float16* __restrict__ Ksi)
{
    const size_t idx = (size_t)blockIdx.x * 256 + threadIdx.x;  // over NB*SS*ROW/4 float4
    float4 kv = ((const float4*)K)[idx];
    v4h srh = { (_Float16)kv.x, (_Float16)(-kv.y),
                (_Float16)kv.z, (_Float16)(-kv.w) };            // (kr, -ki)
    v4h sih = { (_Float16)kv.y, (_Float16)kv.x,
                (_Float16)kv.w, (_Float16)kv.z };               // (ki,  kr)
    *(v4h*)&Ksr[idx * 4] = srh;
    *(v4h*)&Ksi[idx * 4] = sih;
}

// ---- pre-pass 2: V (f32 [b][s][d][comp]) -> f16 transposed [b][comp][d][s]
__global__ __launch_bounds__(256)
void prep_v_kernel(const float* __restrict__ V, _Float16* __restrict__ VT)
{
    const int id   = blockIdx.x * 256 + threadIdx.x;   // over NB*2*64*(SS/8)
    const int sblk = id & (SS / 8 - 1);
    int rest = id >> 9;                                // SS/8 == 512
    const int d    = rest & 63;  rest >>= 6;
    const int comp = rest & 1;
    const int b    = rest >> 1;
    const int s0   = sblk * 8;
    v8h out;
    #pragma unroll
    for (int q = 0; q < 8; ++q)
        out[q] = (_Float16)V[((size_t)b * SS + s0 + q) * ROW + 2 * d + comp];
    *(v8h*)&VT[(((size_t)b * 2 + comp) * 64 + d) * SS + s0] = out;
}

__global__ __launch_bounds__(256)
void cdpa_flash_kernel(const float* __restrict__ Q,
                       const _Float16* __restrict__ Ksr,
                       const _Float16* __restrict__ Ksi,
                       const _Float16* __restrict__ VT,
                       float* __restrict__ O)
{
    __shared__ _Float16 ldsKsr[KT * ROW];          // 32 KB
    __shared__ _Float16 ldsKsi[KT * ROW];          // 32 KB
    __shared__ _Float16 ldsVT[2 * 64 * KT];        // 32 KB
    __shared__ _Float16 ldsP[WAVES][2][16 * KT];   // 64 KB

    const int tid  = threadIdx.x;
    const int w    = tid >> 5;
    const int lane = tid & 31;
    const int m    = lane & 15;   // row (A) / col (B,C) index
    const int h    = lane >> 4;   // half-wave selector

    const int blocksPerBatch = SS / (WAVES * QW);  // 32
    const int b  = blockIdx.x / blocksPerBatch;
    const int qt = blockIdx.x % blocksPerBatch;
    const int q0 = qt * (WAVES * QW) + w * QW;

    // Q A-operands. Scale = 1/sqrt(D) * log2(e): scores in log2 domain.
    const float qscale = 0.125f * 1.44269504088896340736f;
    v16h aq[4];
    {
        const float* qrow = Q + ((size_t)b * SS + q0 + m) * ROW;
        #pragma unroll
        for (int c = 0; c < 4; ++c) {
            #pragma unroll
            for (int j = 0; j < 8; ++j) {
                aq[c][j]     = (_Float16)(qrow[32*c +      8*h + j] * qscale);
                aq[c][8 + j] = (_Float16)(qrow[32*c + 16 + 8*h + j] * qscale);
            }
        }
    }

    // all-ones B operand: rowsum(P) = P @ ones on the matrix pipe
    v16h ones;
    #pragma unroll
    for (int j = 0; j < 16; ++j) ones[j] = (_Float16)1.0f;

    v8f acc[2][2][4];
    #pragma unroll
    for (int mp = 0; mp < 2; ++mp)
        #pragma unroll
        for (int cp = 0; cp < 2; ++cp)
            #pragma unroll
            for (int g = 0; g < 4; ++g)
                #pragma unroll
                for (int j = 0; j < 8; ++j) acc[mp][cp][g][j] = 0.f;

    float mS[2][8], lS[2][8];
    #pragma unroll
    for (int mp = 0; mp < 2; ++mp)
        #pragma unroll
        for (int j = 0; j < 8; ++j) { mS[mp][j] = -__builtin_inff(); lS[mp][j] = 0.f; }

    const _Float16* vtBase = VT + (size_t)b * 2 * 64 * SS;   // rows stride SS

    for (int kt = 0; kt < NTILE; ++kt) {
        __syncthreads();
        // ---- staging: pure b128 vector copies of pre-converted f16 tiles
        {
            const v8h* gsr = (const v8h*)(Ksr + ((size_t)b * SS + (size_t)kt * KT) * ROW);
            const v8h* gsi = (const v8h*)(Ksi + ((size_t)b * SS + (size_t)kt * KT) * ROW);
            #pragma unroll
            for (int i = 0; i < 8; ++i) {          // 2048 v8h units per 32 KB tile
                const int j = tid + 256 * i;
                ((v8h*)ldsKsr)[j] = gsr[j];
                ((v8h*)ldsKsi)[j] = gsi[j];
            }
            const _Float16* gvt = vtBase + (size_t)kt * KT;
            #pragma unroll
            for (int i = 0; i < 8; ++i) {          // 128 rows x 16 units of 8 halves
                const int j = tid + 256 * i;
                const int r = j >> 4, c = j & 15;
                *(v8h*)&ldsVT[r * KT + c * 8] = *(const v8h*)&gvt[(size_t)r * SS + c * 8];
            }
        }
        __syncthreads();

        // ---- prefetch next tile into near caches, overlapped with compute
        if (kt + 1 < NTILE) {
            const char* nsr = (const char*)(Ksr + ((size_t)b * SS + (size_t)(kt + 1) * KT) * ROW);
            const char* nsi = (const char*)(Ksi + ((size_t)b * SS + (size_t)(kt + 1) * KT) * ROW);
            #pragma unroll
            for (int i = 0; i < 2; ++i) {          // 256 thr * 64B * 2 = 32 KB each
                __builtin_prefetch(nsr + tid * 64 + i * 16384, 0, 3);
                __builtin_prefetch(nsi + tid * 64 + i * 16384, 0, 3);
            }
            const char* nvt = (const char*)(vtBase + (size_t)(kt + 1) * KT);
            __builtin_prefetch(nvt + (size_t)(tid >> 1) * (SS * 2) + (tid & 1) * 128, 0, 3);
        }

        // ---- per map: scores (32 WMMAs), online max update, P store
        #pragma unroll
        for (int mp = 0; mp < 2; ++mp) {
            const _Float16* kb = mp ? ldsKsi : ldsKsr;
            v8f sAcc[8];
            #pragma unroll
            for (int g = 0; g < 8; ++g) {
                v8f s = {0.f,0.f,0.f,0.f,0.f,0.f,0.f,0.f};
                const _Float16* krow = &kb[(16 * g + m) * ROW];
                #pragma unroll
                for (int c = 0; c < 4; ++c)
                    s = wmma_f16(aq[c], ld_op(&krow[32 * c + 8 * h]), s);
                sAcc[g] = s;
            }

            _Float16* pM = &ldsP[w][mp][0];
            #pragma unroll
            for (int j = 0; j < 8; ++j) {
                float tm = fmaxf(fmaxf(fmaxf(sAcc[0][j], sAcc[1][j]), fmaxf(sAcc[2][j], sAcc[3][j])),
                                 fmaxf(fmaxf(sAcc[4][j], sAcc[5][j]), fmaxf(sAcc[6][j], sAcc[7][j])));
                #pragma unroll
                for (int msk = 1; msk < 16; msk <<= 1) tm = fmaxf(tm, __shfl_xor(tm, msk, 32));
                float mn = fmaxf(mS[mp][j], tm);
                float f  = fast_exp2(mS[mp][j] - mn);
                mS[mp][j] = mn;
                lS[mp][j] *= f;                  // row-sum added after P@ones
                #pragma unroll
                for (int cp = 0; cp < 2; ++cp)
                    #pragma unroll
                    for (int g = 0; g < 4; ++g) acc[mp][cp][g][j] *= f;
                const int row = j + 8 * h;       // C-layout row for this lane/vgpr
                #pragma unroll
                for (int g = 0; g < 8; ++g)
                    pM[row * KT + 16 * g + m] = (_Float16)fast_exp2(sAcc[g][j] - mn);
            }
        }

        // ---- P @ V and P @ ones: kc-outer keeps only 2 P operands live
        v8f lsum[2];
        #pragma unroll
        for (int mp = 0; mp < 2; ++mp)
            #pragma unroll
            for (int j = 0; j < 8; ++j) lsum[mp][j] = 0.f;

        const _Float16* pM0 = &ldsP[w][0][0];
        const _Float16* pM1 = &ldsP[w][1][0];
        #pragma unroll
        for (int kc = 0; kc < 4; ++kc) {
            v16h pA = ld_op(&pM0[m * KT + 32 * kc + 8 * h]);
            v16h pB = ld_op(&pM1[m * KT + 32 * kc + 8 * h]);
            lsum[0] = wmma_f16(pA, ones, lsum[0]);
            lsum[1] = wmma_f16(pB, ones, lsum[1]);
            #pragma unroll
            for (int cp = 0; cp < 2; ++cp) {
                #pragma unroll
                for (int g4 = 0; g4 < 4; ++g4) {
                    const int d = 16 * g4 + m;
                    v16h vb = ld_op(&ldsVT[(cp * 64 + d) * KT + 32 * kc + 8 * h]);
                    acc[0][cp][g4] = wmma_f16(pA, vb, acc[0][cp][g4]);
                    acc[1][cp][g4] = wmma_f16(pB, vb, acc[1][cp][g4]);
                }
            }
        }
        #pragma unroll
        for (int mp = 0; mp < 2; ++mp)
            #pragma unroll
            for (int j = 0; j < 8; ++j) lS[mp][j] += lsum[mp][j];
    }

    // ---- normalize per term and emit interleaved complex output
    #pragma unroll
    for (int j = 0; j < 8; ++j) {
        float ir = 1.0f / lS[0][j];
        float ii = 1.0f / lS[1][j];
        int row  = q0 + j + 8 * h;
        float* orow = O + ((size_t)b * SS + row) * ROW;
        #pragma unroll
        for (int g4 = 0; g4 < 4; ++g4) {
            int d = 16 * g4 + m;
            float outr = acc[0][0][g4][j] * ir - acc[1][1][g4][j] * ii;
            float outi = acc[0][1][g4][j] * ir + acc[1][0][g4][j] * ii;
            float2 o2  = make_float2(outr, outi);
            *(float2*)(orow + 2 * d) = o2;
        }
    }
}

extern "C" void kernel_launch(void* const* d_in, const int* in_sizes, int n_in,
                              void* d_out, int out_size, void* d_ws, size_t ws_size,
                              hipStream_t stream) {
    const float* Q = (const float*)d_in[0];
    const float* K = (const float*)d_in[1];
    const float* V = (const float*)d_in[2];
    float* O = (float*)d_out;

    // workspace: Ksr (4 MB) | Ksi (4 MB) | VT (4 MB), all f16
    _Float16* wsKsr = (_Float16*)d_ws;
    _Float16* wsKsi = wsKsr + (size_t)NB * SS * ROW;
    _Float16* wsVT  = wsKsi + (size_t)NB * SS * ROW;

    prep_k_kernel<<<dim3(NB * SS * ROW / 4 / 256), dim3(256), 0, stream>>>(K, wsKsr, wsKsi);
    prep_v_kernel<<<dim3(NB * 2 * 64 * (SS / 8) / 256), dim3(256), 0, stream>>>(V, wsVT);
    cdpa_flash_kernel<<<dim3(NB * (SS / (WAVES * QW))), dim3(256), 0, stream>>>(
        Q, wsKsr, wsKsi, wsVT, O);
}
